// LSTM_37778532335717
// MI455X (gfx1250) — matrix operand order repaired and checked by reading, not compiled, not run on hardware
//
#include <hip/hip_runtime.h>
#include <hip/hip_bf16.h>

// ---------------------------------------------------------------------------
// LSTM cell on gfx1250 (MI455X): bf16 WMMA GEMMs, async global->LDS double
// buffering, fused gate epilogue.
//   B=8192, IN=1024, H=1024, OUT=512, K = IN+H = 2048
//   out = [ y_hat (8192x512) | h_new (8192x1024) | c_new (8192x1024) ] fp32
// ---------------------------------------------------------------------------

#define BROWS   8192
#define KDIM    2048
#define HDIM    1024
#define ODIM    512

#define MT      128     // workgroup M tile
#define NT      32      // workgroup N tile (per gate)
#define KT      64      // K staging tile

typedef __attribute__((ext_vector_type(16))) __bf16 v16bf;
typedef __attribute__((ext_vector_type(8)))  float  v8f;
typedef __attribute__((__vector_size__(4 * sizeof(int)))) int v4i;

union FragAB {
    v16bf v;
    uint4 q[2];
};

__device__ __forceinline__ unsigned short f2bf(float f) {
    union { float f; unsigned int u; } cv;
    cv.f = f;
    unsigned int u = cv.u;
    unsigned int r = u + 0x7FFFu + ((u >> 16) & 1u);   // round-to-nearest-even
    return (unsigned short)(r >> 16);
}

__device__ __forceinline__ float sigmoidf_fast(float x) {
    return 1.0f / (1.0f + __expf(-x));
}

// --- gfx1250 async global->LDS copy (ASYNCcnt-tracked) ----------------------

__device__ __forceinline__ void async_copy_b128(void* lds_dst, const void* gsrc) {
#if __has_builtin(__builtin_amdgcn_global_load_async_to_lds_b128)
    __builtin_amdgcn_global_load_async_to_lds_b128(
        (__attribute__((address_space(1))) v4i*)gsrc,
        (__attribute__((address_space(3))) v4i*)lds_dst, 0, 0);
#else
    // LDS byte offset = low 32 bits of the generic shared-space address
    unsigned lds_off = (unsigned)(unsigned long long)lds_dst;
    asm volatile("global_load_async_to_lds_b128 %0, %1, off"
                 :: "v"(lds_off), "v"(gsrc)
                 : "memory");
#endif
}

__device__ __forceinline__ void wait_async_all() {
#if __has_builtin(__builtin_amdgcn_s_wait_asynccnt)
    __builtin_amdgcn_s_wait_asynccnt(0);
#else
    asm volatile("s_wait_asynccnt 0x0" ::: "memory");
#endif
}

// --- conversion kernels -----------------------------------------------------

// combined_bf16[8192][2048] = bf16(concat(x, h))
__global__ __launch_bounds__(256) void k_convert_combined(
    const float* __restrict__ x, const float* __restrict__ h,
    unsigned short* __restrict__ out)
{
    const unsigned idx = blockIdx.x * 256u + threadIdx.x;      // < 8192*2048
    const unsigned row = idx >> 11;
    const unsigned col = idx & 2047u;
    float v = (col < 1024u) ? x[(size_t)row * 1024u + col]
                            : h[(size_t)row * 1024u + (col - 1024u)];
    out[idx] = f2bf(v);
}

// Wt[g][n][k] = bf16(W_g[k][n]),  g in {i,f,o,c}, n<1024, k<2048
__global__ __launch_bounds__(256) void k_convert_gatew(
    const float* __restrict__ Wi, const float* __restrict__ Wf,
    const float* __restrict__ Wo, const float* __restrict__ Wc,
    unsigned short* __restrict__ out)
{
    const unsigned idx = blockIdx.x * 256u + threadIdx.x;      // < 4*1024*2048
    const unsigned g   = idx >> 21;
    const unsigned rem = idx & ((1u << 21) - 1u);
    const unsigned n   = rem >> 11;
    const unsigned k   = rem & 2047u;
    const float* W = (g == 0) ? Wi : (g == 1) ? Wf : (g == 2) ? Wo : Wc;
    out[idx] = f2bf(W[(size_t)k * 1024u + n]);
}

// Wyt[n][k] = bf16(Wy[k][n]), n<512, k<1024
__global__ __launch_bounds__(256) void k_convert_wy(
    const float* __restrict__ Wy, unsigned short* __restrict__ out)
{
    const unsigned idx = blockIdx.x * 256u + threadIdx.x;      // < 512*1024
    const unsigned n = idx >> 10;
    const unsigned k = idx & 1023u;
    out[idx] = f2bf(Wy[(size_t)k * 512u + n]);
}

// --- fused 4-gate GEMM + LSTM epilogue --------------------------------------
// A: combined bf16 [8192][2048] row-major
// Wt: bf16 [4][1024][2048] (n-major, k-contiguous)
// grid: (8192/MT, 1024/NT), block 256 = 8 waves in 4(M) x 2(N) arrangement
__global__ __launch_bounds__(256) void k_lstm_gemm(
    const unsigned short* __restrict__ Abf,
    const unsigned short* __restrict__ Wt,
    const float* __restrict__ bi, const float* __restrict__ bfv,
    const float* __restrict__ bo, const float* __restrict__ bc,
    const float* __restrict__ c_old,
    float* __restrict__ h_out, float* __restrict__ c_out,
    unsigned short* __restrict__ h_bf)
{
    __shared__ __align__(16) unsigned short Al[2][MT * KT];          // 2x16 KB
    __shared__ __align__(16) unsigned short Bl[2][4 * NT * KT];      // 2x16 KB

    const int t    = threadIdx.x;
    const int lane = t & 31;
    const int wave = t >> 5;
    const int wm   = wave & 3;       // rows wm*32
    const int wn   = wave >> 2;      // cols wn*16
    const int m0   = blockIdx.x * MT;
    const int n0   = blockIdx.y * NT;
    const int half = lane >> 4;
    const int l16  = lane & 15;

    // Async-stage one K tile into LDS buffer `buf` (each thread: 8 x b128)
    auto stage = [&](int k0, int buf) {
        #pragma unroll
        for (int i = 0; i < 4; ++i) {
            int c   = t + 256 * i;
            int row = c >> 3;
            int kc  = (c & 7) << 3;
            async_copy_b128(&Al[buf][row * KT + kc],
                            &Abf[(size_t)(m0 + row) * KDIM + k0 + kc]);
        }
        #pragma unroll
        for (int i = 0; i < 4; ++i) {
            int c   = t + 256 * i;
            int g   = c >> 8;
            int rem = c & 255;
            int nn  = rem >> 3;
            int kc  = (rem & 7) << 3;
            async_copy_b128(&Bl[buf][(g * NT + nn) * KT + kc],
                            &Wt[((size_t)g * HDIM + (n0 + nn)) * KDIM + k0 + kc]);
        }
    };

    v8f acc[4][2];
    for (int g = 0; g < 4; ++g)
        for (int mf = 0; mf < 2; ++mf)
            acc[g][mf] = {};

    stage(0, 0);

    int buf = 0;
    for (int k0 = 0; k0 < KDIM; k0 += KT) {
        wait_async_all();     // this wave's portion of LDS tile is written
        __syncthreads();      // whole tile visible; prev buffer fully consumed
        if (k0 + KT < KDIM)
            stage(k0 + KT, buf ^ 1);   // overlap next tile with compute

        const unsigned short* Ab = Al[buf];
        const unsigned short* Bb = Bl[buf];

        #pragma unroll
        for (int ks = 0; ks < KT; ks += 32) {
            // A fragments: 16x32 bf16; lane half selects K 0-7 | 8-15 (+16)
            FragAB a[2];
            #pragma unroll
            for (int mf = 0; mf < 2; ++mf) {
                const unsigned short* ap = &Ab[(wm * 32 + mf * 16 + l16) * KT + ks];
                a[mf].q[0] = *(const uint4*)&ap[half * 8];
                a[mf].q[1] = *(const uint4*)&ap[16 + half * 8];
            }
            // Preload all 4 gate B fragments (distinct regs -> overlap loads)
            FragAB b[4];
            #pragma unroll
            for (int g = 0; g < 4; ++g) {
                const unsigned short* bp = &Bb[(g * NT + wn * 16 + l16) * KT + ks];
                b[g].q[0] = *(const uint4*)&bp[half * 8];
                b[g].q[1] = *(const uint4*)&bp[16 + half * 8];
            }
            #pragma unroll
            for (int g = 0; g < 4; ++g) {
                #pragma unroll
                for (int mf = 0; mf < 2; ++mf) {
                    acc[g][mf] = __builtin_amdgcn_wmma_f32_16x16x32_bf16(
                        false, a[mf].v, false, b[g].v, (short)0, acc[g][mf],
                        false, false);
                }
            }
        }
        buf ^= 1;
    }

    // Epilogue: per-lane column n; C/D layout rows = vgpr + 8*half
    const int ncol = n0 + wn * 16 + l16;
    const float vbi = bi[ncol], vbf = bfv[ncol], vbo = bo[ncol], vbc = bc[ncol];
    #pragma unroll
    for (int mf = 0; mf < 2; ++mf) {
        const int rbase = m0 + wm * 32 + mf * 16 + half * 8;
        #pragma unroll
        for (int r = 0; r < 8; ++r) {
            const int row = rbase + r;
            const float gi = acc[0][mf][r] + vbi;
            const float gf = acc[1][mf][r] + vbf;
            const float go = acc[2][mf][r] + vbo;
            const float gc = acc[3][mf][r] + vbc;
            const float it = sigmoidf_fast(gi);
            const float ft = sigmoidf_fast(gf);
            const float ot = sigmoidf_fast(go);
            const float ct = tanhf(gc);
            const size_t idx = (size_t)row * HDIM + ncol;
            const float cn = ft * c_old[idx] + it * ct;
            const float hn = ot * tanhf(cn);
            c_out[idx] = cn;
            h_out[idx] = hn;
            h_bf[idx]  = f2bf(hn);
        }
    }
}

// --- output GEMM: y = h_new @ Wy + by ---------------------------------------
// A: h_new bf16 [8192][1024]; Wyt bf16 [512][1024]; grid (8192/MT, 512/NT)
__global__ __launch_bounds__(256) void k_out_gemm(
    const unsigned short* __restrict__ Abf,
    const unsigned short* __restrict__ Wyt,
    const float* __restrict__ by,
    float* __restrict__ y)
{
    __shared__ __align__(16) unsigned short Al[2][MT * KT];      // 2x16 KB
    __shared__ __align__(16) unsigned short Bl[2][NT * KT];      // 2x4 KB

    const int t    = threadIdx.x;
    const int lane = t & 31;
    const int wave = t >> 5;
    const int wm   = wave & 3;
    const int wn   = wave >> 2;
    const int m0   = blockIdx.x * MT;
    const int n0   = blockIdx.y * NT;
    const int half = lane >> 4;
    const int l16  = lane & 15;

    auto stage = [&](int k0, int buf) {
        #pragma unroll
        for (int i = 0; i < 4; ++i) {
            int c   = t + 256 * i;
            int row = c >> 3;
            int kc  = (c & 7) << 3;
            async_copy_b128(&Al[buf][row * KT + kc],
                            &Abf[(size_t)(m0 + row) * HDIM + k0 + kc]);
        }
        {
            int nn = t >> 3;
            int kc = (t & 7) << 3;
            async_copy_b128(&Bl[buf][nn * KT + kc],
                            &Wyt[(size_t)(n0 + nn) * HDIM + k0 + kc]);
        }
    };

    v8f acc[2];
    acc[0] = {};
    acc[1] = {};

    stage(0, 0);

    int buf = 0;
    for (int k0 = 0; k0 < HDIM; k0 += KT) {
        wait_async_all();
        __syncthreads();
        if (k0 + KT < HDIM)
            stage(k0 + KT, buf ^ 1);

        const unsigned short* Ab = Al[buf];
        const unsigned short* Bb = Bl[buf];

        #pragma unroll
        for (int ks = 0; ks < KT; ks += 32) {
            FragAB a[2];
            #pragma unroll
            for (int mf = 0; mf < 2; ++mf) {
                const unsigned short* ap = &Ab[(wm * 32 + mf * 16 + l16) * KT + ks];
                a[mf].q[0] = *(const uint4*)&ap[half * 8];
                a[mf].q[1] = *(const uint4*)&ap[16 + half * 8];
            }
            FragAB b;
            const unsigned short* bp = &Bb[(wn * 16 + l16) * KT + ks];
            b.q[0] = *(const uint4*)&bp[half * 8];
            b.q[1] = *(const uint4*)&bp[16 + half * 8];
            #pragma unroll
            for (int mf = 0; mf < 2; ++mf) {
                acc[mf] = __builtin_amdgcn_wmma_f32_16x16x32_bf16(
                    false, a[mf].v, false, b.v, (short)0, acc[mf], false, false);
            }
        }
        buf ^= 1;
    }

    const int ncol = n0 + wn * 16 + l16;
    const float vb = by[ncol];
    #pragma unroll
    for (int mf = 0; mf < 2; ++mf) {
        const int rbase = m0 + wm * 32 + mf * 16 + half * 8;
        #pragma unroll
        for (int r = 0; r < 8; ++r) {
            y[(size_t)(rbase + r) * ODIM + ncol] = acc[mf][r] + vb;
        }
    }
}

// ---------------------------------------------------------------------------

extern "C" void kernel_launch(void* const* d_in, const int* in_sizes, int n_in,
                              void* d_out, int out_size, void* d_ws, size_t ws_size,
                              hipStream_t stream) {
    const float* x  = (const float*)d_in[0];
    const float* h  = (const float*)d_in[1];
    const float* c  = (const float*)d_in[2];
    const float* Wi = (const float*)d_in[3];
    const float* bi = (const float*)d_in[4];
    const float* Wf = (const float*)d_in[5];
    const float* bf = (const float*)d_in[6];
    const float* Wo = (const float*)d_in[7];
    const float* bo = (const float*)d_in[8];
    const float* Wc = (const float*)d_in[9];
    const float* bc = (const float*)d_in[10];
    const float* Wy = (const float*)d_in[11];
    const float* by = (const float*)d_in[12];

    float* y_out = (float*)d_out;                               // [8192][512]
    float* h_out = y_out + (size_t)BROWS * ODIM;                // [8192][1024]
    float* c_out = h_out + (size_t)BROWS * HDIM;                // [8192][1024]

    // Workspace layout (bf16 staging buffers)
    char* ws = (char*)d_ws;
    unsigned short* combined = (unsigned short*)(ws);                       // 33.5 MB
    unsigned short* Wt       = (unsigned short*)(ws + (size_t)33554432);    // 16.8 MB
    unsigned short* Wyt      = (unsigned short*)(ws + (size_t)50331648);    //  1.0 MB
    unsigned short* hbf      = (unsigned short*)(ws + (size_t)51380224);    // 16.8 MB

    k_convert_combined<<<dim3(65536), dim3(256), 0, stream>>>(x, h, combined);
    k_convert_gatew<<<dim3(32768), dim3(256), 0, stream>>>(Wi, Wf, Wo, Wc, Wt);
    k_convert_wy<<<dim3(2048), dim3(256), 0, stream>>>(Wy, Wyt);

    dim3 g1(BROWS / MT, HDIM / NT);   // 64 x 32
    k_lstm_gemm<<<g1, dim3(256), 0, stream>>>(combined, Wt, bi, bf, bo, bc, c,
                                              h_out, c_out, hbf);

    dim3 g2(BROWS / MT, ODIM / NT);   // 64 x 16
    k_out_gemm<<<g2, dim3(256), 0, stream>>>(hbf, Wyt, by, y_out);
}
